// Block_84911503442222
// MI455X (gfx1250) — matrix operand order repaired and checked by reading, not compiled, not run on hardware
//
#include <hip/hip_runtime.h>
#include <math.h>

// ---------------------------------------------------------------------------
// Bidirectional Mamba block for MI455X (gfx1250): wave32, bf16 WMMA,
// TDM (tensor_load_to_lds) staging for channel-major GEMM operands.
//
// B=2, D_MODEL=1024, L=2048, D_INNER=2048, D_STATE=16, DT_RANK=64, D_CONV=4.
// ---------------------------------------------------------------------------

#define LSEQ   2048
#define DMODEL 1024
#define DINNER 2048
#define DSTATE 16
#define DTRANK 64
#define NBATCH 2
#define NTOK   (NBATCH * LSEQ)          // 4096
#define KC     64                       // TDM chunk: 64 k-rows x 16 cols (bf16)

typedef __attribute__((ext_vector_type(16))) __bf16    v16bf;
typedef __attribute__((ext_vector_type(8)))  float     v8f;
typedef __attribute__((ext_vector_type(4)))  unsigned  u32x4;
typedef __attribute__((ext_vector_type(8)))  int       i32x8;
typedef __attribute__((ext_vector_type(4)))  int       i32x4;

// float -> bf16 (RNE) as raw u16, pure bit math
__device__ __forceinline__ unsigned short f2bfu(float f) {
    union { float f; unsigned u; } a; a.f = f;
    unsigned u = a.u;
    u += 0x7fffu + ((u >> 16) & 1u);
    return (unsigned short)(u >> 16);
}
__device__ __forceinline__ __bf16 u2bf(unsigned short s) {
    union { unsigned short u; __bf16 b; } c; c.u = s; return c.b;
}
__device__ __forceinline__ float siluf(float x) { return x / (1.f + expf(-x)); }
__device__ __forceinline__ float softplusf(float x) {
    return (x > 20.f) ? x : log1pf(expf(x));
}

// ---------------------------------------------------------------------------
// f32 -> bf16 elementwise (weights)
// ---------------------------------------------------------------------------
__global__ void cvt_bf16_kernel(const float* __restrict__ src,
                                unsigned short* __restrict__ dst, int n) {
    int i = blockIdx.x * 256 + threadIdx.x;
    if (i < n) dst[i] = f2bfu(src[i]);
}

// ---------------------------------------------------------------------------
// Prelude: r = hidden + residual -> output #2 (original (B,D,L) layout);
// LayerNorm over D -> hn_bf token-major [m][d] in bf16 (GEMM1 B operand).
// ---------------------------------------------------------------------------
__global__ void prelude_kernel(const float* __restrict__ hidden,
                               const float* __restrict__ residual,
                               const float* __restrict__ gamma,
                               const float* __restrict__ beta,
                               unsigned short* __restrict__ hn_bf,
                               float* __restrict__ out_r) {
    __shared__ float ssum[8][32];
    __shared__ float ssq[8][32];
    int l_off = threadIdx.x & 31;
    int d_off = threadIdx.x >> 5;
    int t0 = blockIdx.x * 32;
    int b  = t0 >> 11;
    int l  = (t0 & 2047) + l_off;

    float s = 0.f, q = 0.f;
    for (int d = d_off; d < DMODEL; d += 8) {
        size_t gi = ((size_t)(b * DMODEL + d)) * LSEQ + l;
        float r = hidden[gi] + residual[gi];
        out_r[gi] = r;
        s += r; q += r * r;
    }
    ssum[d_off][l_off] = s;
    ssq[d_off][l_off]  = q;
    __syncthreads();
    float ts = 0.f, tq = 0.f;
    for (int j = 0; j < 8; ++j) { ts += ssum[j][l_off]; tq += ssq[j][l_off]; }
    float mu  = ts * (1.f / DMODEL);
    float var = tq * (1.f / DMODEL) - mu * mu;
    float rs  = rsqrtf(var + 1e-5f);

    size_t m = (size_t)b * LSEQ + l;
    for (int d = d_off; d < DMODEL; d += 8) {
        size_t gi = ((size_t)(b * DMODEL + d)) * LSEQ + l;
        float r = hidden[gi] + residual[gi];
        hn_bf[m * DMODEL + d] = f2bfu((r - mu) * rs * gamma[d] + beta[d]);
    }
}

// ---------------------------------------------------------------------------
// WMMA GEMM (bf16 operands, f32 accum).  Out[i,n] = sum_k W[i,k]*B(k,n).
//   BMODE 0: B(k,n) = X[n*K + k]         (token-major, contiguous per lane)
//   BMODE 1: B(k,n) = X[((n>>11)*K + k)*2048 + (n&2047)]  (channel-major)
//            -> staged into LDS with TDM tensor_load_to_lds, double-buffered
//               KC-row chunks, s_wait_tensorcnt sync.
//   EPI 0: store; 1: +bias; 2: softplus(x+bias).  WRDT: also emit bf16 copy
//   of rows i<64 (the dt slice consumed by the dt_proj GEMM).
//   MT: i-subtiles per wave (exact: I == gI * MT * 16), so there are NO
//   runtime guards around the WMMAs -> no EXEC masking, clean scheduling.
// One wave = (MT*16)(I) x 16(N) strip: MT accumulators share each B fragment.
// A/B fragments use contiguous k (element i: k = half*16 + i) in both operands.
// ---------------------------------------------------------------------------
template <int BMODE, int EPI, int WRDT, int MT>
__global__ __launch_bounds__(256)
void gemm_bf16_kernel(const unsigned short* __restrict__ W,
                      const unsigned short* __restrict__ X,
                      float* __restrict__ Out,
                      const float* __restrict__ bias,
                      unsigned short* __restrict__ dtbf,
                      int I, int K, int N) {
    extern __shared__ unsigned short ldsB[];   // [8 waves][2 bufs][KC*16]
    int wave = threadIdx.x >> 5;
    int lane = threadIdx.x & 31;
    int tilesN = N >> 4;
    int gI = I / (16 * MT);                    // exact by construction
    int tile = blockIdx.x * 8 + wave;
    if (tile >= gI * tilesN) return;           // whole-wave exit, EXEC stays full
    int ig = tile / tilesN;
    int nt = tile - ig * tilesN;
    int i0 = ig * (16 * MT);
    int n0 = nt << 4;

    int half   = lane >> 4;
    int lane15 = lane & 15;
    int n  = n0 + lane15;
    int nb = n >> 11;
    int nl = n & 2047;

    const unsigned short* Abase = W + (size_t)(i0 + lane15) * K + half * 16;

    const unsigned short* Bptr = nullptr;      // BMODE 0
    const unsigned short* Bg   = nullptr;      // BMODE 1 (uniform tile base)
    if (BMODE == 0) {
        Bptr = X + (size_t)n * K + half * 16;
    } else {
        Bg = X + ((size_t)(n0 >> 11) * K) * LSEQ + (n0 & 2047);
    }

    v8f acc[MT];
    #pragma unroll
    for (int t = 0; t < MT; ++t)
        for (int r = 0; r < 8; ++r) acc[t][r] = 0.f;

    union U4 { uint4 v; unsigned short s[8]; };

    // A fragment: element i holds k = k0 + half*16 + i
    auto loadA = [&](int t, int k0) -> v16bf {
        const unsigned short* row = Abase + (size_t)(t * 16) * K;
        U4 x0, x1;
        x0.v = *reinterpret_cast<const uint4*>(row + k0);       // k0 .. +7
        x1.v = *reinterpret_cast<const uint4*>(row + k0 + 8);   // k0+8 .. +15
        v16bf a;
        #pragma unroll
        for (int i = 0; i < 8; ++i) { a[i] = u2bf(x0.s[i]); a[i + 8] = u2bf(x1.s[i]); }
        return a;
    };

    auto mma_step = [&](const v16bf& bf, int kg) {
        v16bf a[MT];
        #pragma unroll
        for (int t = 0; t < MT; ++t) a[t] = loadA(t, kg);   // all loads up front
        #pragma unroll
        for (int t = 0; t < MT; ++t)
            acc[t] = __builtin_amdgcn_wmma_f32_16x16x32_bf16(
                false, a[t], false, bf, (short)0, acc[t], false, false);
    };

    if (BMODE == 0) {
        for (int k0 = 0; k0 < K; k0 += 32) {
            if (k0 + 64 < K) __builtin_prefetch(Bptr + k0 + 64, 0, 1);
            U4 b0, b1;                          // element i: k = k0 + half*16 + i
            b0.v = *reinterpret_cast<const uint4*>(Bptr + k0);
            b1.v = *reinterpret_cast<const uint4*>(Bptr + k0 + 8);
            v16bf bf;
            #pragma unroll
            for (int i = 0; i < 8; ++i) { bf[i] = u2bf(b0.s[i]); bf[i + 8] = u2bf(b1.s[i]); }
            mma_step(bf, k0);
        }
    } else {
        // ---- TDM-staged channel-major B ----
        auto issue_tdm = [&](int chunk, int buf) {
            unsigned ldsoff = (unsigned)(size_t)&ldsB[(size_t)(wave * 2 + buf) * (KC * 16)];
            unsigned long long ga =
                (unsigned long long)(size_t)(Bg + (size_t)chunk * KC * LSEQ);
            unsigned lo = (unsigned)__builtin_amdgcn_readfirstlane((int)(ga & 0xffffffffu));
            unsigned hi = (unsigned)__builtin_amdgcn_readfirstlane((int)(ga >> 32));
            unsigned lw = (unsigned)__builtin_amdgcn_readfirstlane((int)ldsoff);
            u32x4 g0; g0[0] = 1u;               // count=1, user descriptor
            g0[1] = lw;                         // lds_addr
            g0[2] = lo;                         // global_addr[31:0]
            g0[3] = (hi & 0x01ffffffu) | (2u << 30);  // addr[56:32] | type=2
            i32x8 g1;
            g1[0] = (int)(1u << 16);            // data_size = 2 bytes
            g1[1] = (int)(2048u << 16);         // tensor_dim0 = 2048 (row length)
            g1[2] = (int)((unsigned)K << 16);   // tensor_dim1 = K rows
            g1[3] = (int)(16u << 16);           // tile_dim0 = 16 cols
            g1[4] = KC;                         // tile_dim1 = KC rows (tile_dim2=0)
            g1[5] = 2048;                       // tensor_dim0_stride
            g1[6] = 0; g1[7] = 0;
            i32x4 z4 = {0, 0, 0, 0};
            i32x8 z8 = {0, 0, 0, 0, 0, 0, 0, 0};
#if __clang_major__ >= 23
            __builtin_amdgcn_tensor_load_to_lds(g0, g1, z4, z4, z8, 0);
            (void)z8;
#else
            __builtin_amdgcn_tensor_load_to_lds(g0, g1, z4, z4, 0);
            (void)z8;
#endif
        };

        int nchunks = K / KC;
        issue_tdm(0, 0);
        for (int c = 0; c < nchunks; ++c) {
            if (c + 1 < nchunks) {
                issue_tdm(c + 1, (c + 1) & 1);
                __builtin_amdgcn_s_wait_tensorcnt(1);   // chunk c landed
            } else {
                __builtin_amdgcn_s_wait_tensorcnt(0);
            }
            const unsigned short* buf = &ldsB[(size_t)(wave * 2 + (c & 1)) * (KC * 16)];
            for (int k0r = 0; k0r < KC; k0r += 32) {
                v16bf bf;                        // element i: k = k0r + half*16 + i
                #pragma unroll
                for (int i = 0; i < 16; ++i)
                    bf[i] = u2bf(buf[(size_t)(k0r + half * 16 + i) * 16 + lane15]);
                mma_step(bf, c * KC + k0r);
            }
        }
    }

    // ---- epilogue ----
    #pragma unroll
    for (int t = 0; t < MT; ++t) {
        #pragma unroll
        for (int r = 0; r < 8; ++r) {
            int i = i0 + t * 16 + r + half * 8;
            float v = acc[t][r];
            if (EPI == 1) v += bias[i];
            if (EPI == 2) v = softplusf(v + bias[i]);
            Out[((size_t)nb * I + i) * LSEQ + nl] = v;
            if (WRDT && i < DTRANK)
                dtbf[((size_t)nb * DTRANK + i) * LSEQ + nl] = f2bfu(v);
        }
    }
}

// ---------------------------------------------------------------------------
// Depthwise causal conv(4) + SiLU, direction coordinates.  Emits f32 (scan u)
// and bf16 (x_proj GEMM operand).
// ---------------------------------------------------------------------------
__global__ void conv_silu_kernel(const float* __restrict__ xz,
                                 const float* __restrict__ w,
                                 const float* __restrict__ bias,
                                 float* __restrict__ xc,
                                 unsigned short* __restrict__ xc_bf,
                                 int rev) {
    size_t idx = (size_t)blockIdx.x * 256 + threadIdx.x;
    int l = (int)(idx & 2047);
    size_t bd = idx >> 11;
    int d = (int)(bd & 2047);
    int b = (int)(bd >> 11);
    const float* xs = xz + ((size_t)(b * 2 * DINNER + d)) * LSEQ;
    float acc = bias[d];
    #pragma unroll
    for (int j = 0; j < 4; ++j) {
        int li = l - 3 + j;
        if (li >= 0) {
            int ls = rev ? (LSEQ - 1 - li) : li;
            acc += w[d * 4 + j] * xs[ls];
        }
    }
    float v = siluf(acc);
    size_t oi = ((size_t)(b * DINNER + d)) * LSEQ + l;
    xc[oi] = v;
    xc_bf[oi] = f2bfu(v);
}

// ---------------------------------------------------------------------------
// Selective scan: lane-per-(channel,state), 16-lane groups, shfl_xor reduce.
// float4-vectorized streaming loads on the serial l-loop.  Backward pass
// (rev=1, accum=1) fuses y = y_f + rev(y_b), SiLU(z) gating, and emits the
// bf16 copy consumed by the out_proj GEMM.
// ---------------------------------------------------------------------------
__global__ void scan_kernel(const float* __restrict__ delta,
                            const float* __restrict__ xc,
                            const float* __restrict__ xdbl,
                            const float* __restrict__ A_log,
                            const float* __restrict__ Dp,
                            const float* __restrict__ xz,
                            float* __restrict__ ysum,
                            unsigned short* __restrict__ ybf,
                            int rev, int accum) {
    int gid = blockIdx.x * blockDim.x + threadIdx.x;
    int ns  = gid & 15;
    int ch  = gid >> 4;
    int d   = ch & (DINNER - 1);
    int b   = ch >> 11;

    float a  = -expf(A_log[d * DSTATE + ns]);
    float dp = Dp[d];
    const float* dtp = delta + ((size_t)(b * DINNER + d)) * LSEQ;
    const float* up  = xc    + ((size_t)(b * DINNER + d)) * LSEQ;
    const float* Bp  = xdbl  + ((size_t)(b * 96) + DTRANK + ns) * LSEQ;
    const float* Cp  = Bp + (size_t)DSTATE * LSEQ;
    const float* zp  = xz + ((size_t)(b * 2 * DINNER) + DINNER + d) * LSEQ;
    float* yp = ysum + ((size_t)(b * DINNER + d)) * LSEQ;
    unsigned short* ybp = ybf + ((size_t)(b * DINNER + d)) * LSEQ;

    const float LOG2E = 1.4426950408889634f;
    float h = 0.f;
    for (int l4 = 0; l4 < LSEQ; l4 += 4) {
        float4 dtv = *reinterpret_cast<const float4*>(dtp + l4);
        float4 uv  = *reinterpret_cast<const float4*>(up + l4);
        float4 bv  = *reinterpret_cast<const float4*>(Bp + l4);
        float4 cv  = *reinterpret_cast<const float4*>(Cp + l4);
        float dts[4] = {dtv.x, dtv.y, dtv.z, dtv.w};
        float us [4] = {uv.x, uv.y, uv.z, uv.w};
        float bs [4] = {bv.x, bv.y, bv.z, bv.w};
        float cs [4] = {cv.x, cv.y, cv.z, cv.w};
        #pragma unroll
        for (int j = 0; j < 4; ++j) {
            float dt = dts[j], u = us[j];
            float e = exp2f(LOG2E * dt * a);     // exp(dt * A_n)
            h = e * h + (dt * u) * bs[j];
            float p = h * cs[j];
            p += __shfl_xor(p, 1, 16);
            p += __shfl_xor(p, 2, 16);
            p += __shfl_xor(p, 4, 16);
            p += __shfl_xor(p, 8, 16);
            if (ns == 0) {
                int lo = rev ? (LSEQ - 1 - (l4 + j)) : (l4 + j);
                float zv = zp[lo];
                float g  = (p + dp * u) * siluf(zv);
                if (accum) {
                    float tot = yp[lo] + g;
                    yp[lo]  = tot;
                    ybp[lo] = f2bfu(tot);
                } else {
                    yp[lo] = g;
                }
            }
        }
    }
}

// ---------------------------------------------------------------------------
extern "C" void kernel_launch(void* const* d_in, const int* in_sizes, int n_in,
                              void* d_out, int out_size, void* d_ws, size_t ws_size,
                              hipStream_t stream) {
    (void)in_sizes; (void)n_in; (void)out_size; (void)ws_size;
    const float* hidden    = (const float*)d_in[0];
    const float* residual  = (const float*)d_in[1];
    const float* gamma     = (const float*)d_in[2];
    const float* beta      = (const float*)d_in[3];
    const float* in_proj_w = (const float*)d_in[4];
    const float* conv_w    = (const float*)d_in[5];
    const float* conv_b    = (const float*)d_in[6];
    const float* xproj_w   = (const float*)d_in[7];
    const float* dtproj_w  = (const float*)d_in[8];
    const float* dtproj_b  = (const float*)d_in[9];
    const float* A_log     = (const float*)d_in[10];
    const float* Dp        = (const float*)d_in[11];
    const float* conv_w_b  = (const float*)d_in[12];
    const float* conv_b_b  = (const float*)d_in[13];
    const float* xproj_w_b = (const float*)d_in[14];
    const float* dtproj_w_b= (const float*)d_in[15];
    const float* dtproj_b_b= (const float*)d_in[16];
    const float* A_b_log   = (const float*)d_in[17];
    const float* Dp_b      = (const float*)d_in[18];
    const float* out_proj_w= (const float*)d_in[19];
    const float* out_proj_b= (const float*)d_in[20];

    float* out1 = (float*)d_out;
    float* out2 = out1 + (size_t)NBATCH * DMODEL * LSEQ;

    // ---- workspace carve-up (16-byte aligned slabs) ----
    char* wsb = (char*)d_ws;
    auto alloc = [&](size_t bytes) {
        char* p = wsb;
        wsb += (bytes + 15) & ~(size_t)15;
        return p;
    };
    float*          xz    = (float*)alloc((size_t)NBATCH * 2 * DINNER * LSEQ * 4);
    float*          xc    = (float*)alloc((size_t)NBATCH * DINNER * LSEQ * 4);
    float*          delta = (float*)alloc((size_t)NBATCH * DINNER * LSEQ * 4);
    float*          ysum  = (float*)alloc((size_t)NBATCH * DINNER * LSEQ * 4);
    float*          xdbl  = (float*)alloc((size_t)NBATCH * 96 * LSEQ * 4);
    unsigned short* hn_bf = (unsigned short*)alloc((size_t)NTOK * DMODEL * 2);
    unsigned short* xc_bf = (unsigned short*)alloc((size_t)NBATCH * DINNER * LSEQ * 2);
    unsigned short* ybf   = (unsigned short*)alloc((size_t)NBATCH * DINNER * LSEQ * 2);
    unsigned short* dtbf  = (unsigned short*)alloc((size_t)NBATCH * DTRANK * LSEQ * 2);
    unsigned short* w1bf  = (unsigned short*)alloc((size_t)2 * DINNER * DMODEL * 2);
    unsigned short* w2bf  = (unsigned short*)alloc((size_t)96 * DINNER * 2);
    unsigned short* w3bf  = (unsigned short*)alloc((size_t)DINNER * DTRANK * 2);
    unsigned short* w4bf  = (unsigned short*)alloc((size_t)DMODEL * DINNER * 2);

    const int TDM_LDS = 8 * 2 * KC * 16 * 2;   // 32 KB

    auto cvt = [&](const float* s, unsigned short* dst, int nelem) {
        cvt_bf16_kernel<<<(nelem + 255) / 256, 256, 0, stream>>>(s, dst, nelem);
    };
    auto gblocks = [](int I, int N, int MT) {
        int tiles = (I / (16 * MT)) * (N >> 4);
        return (tiles + 7) / 8;
    };

    // 1) prelude (+ output #2)
    prelude_kernel<<<NTOK / 32, 256, 0, stream>>>(hidden, residual, gamma, beta, hn_bf, out2);

    // 2) xz = in_proj_w * hn^T  (token-major B operand), I=4096 -> MT=4
    cvt(in_proj_w, w1bf, 2 * DINNER * DMODEL);
    gemm_bf16_kernel<0, 0, 0, 4><<<gblocks(2 * DINNER, NTOK, 4), 256, 0, stream>>>(
        w1bf, hn_bf, xz, nullptr, nullptr, 2 * DINNER, DMODEL, NTOK);

    const int convBlocks = (int)(((size_t)NBATCH * DINNER * LSEQ) / 256);
    const int scanBlocks = (NBATCH * DINNER * DSTATE) / 256;

    for (int dir = 0; dir < 2; ++dir) {
        const float* cw  = dir ? conv_w_b    : conv_w;
        const float* cb  = dir ? conv_b_b    : conv_b;
        const float* xw  = dir ? xproj_w_b   : xproj_w;
        const float* dw  = dir ? dtproj_w_b  : dtproj_w;
        const float* db  = dir ? dtproj_b_b  : dtproj_b;
        const float* al  = dir ? A_b_log     : A_log;
        const float* dpp = dir ? Dp_b        : Dp;

        cvt(xw, w2bf, 96 * DINNER);
        cvt(dw, w3bf, DINNER * DTRANK);

        conv_silu_kernel<<<convBlocks, 256, 0, stream>>>(xz, cw, cb, xc, xc_bf, dir);

        // x_proj: I=96 -> MT=2 (6 tiles = 3 groups of 2, exact); TDM-staged B;
        // also emits bf16 dt slice
        gemm_bf16_kernel<1, 0, 1, 2><<<gblocks(96, NTOK, 2), 256, TDM_LDS, stream>>>(
            w2bf, xc_bf, xdbl, nullptr, dtbf, 96, DINNER, NTOK);

        // dt_proj: I=2048 -> MT=4; + softplus epilogue
        gemm_bf16_kernel<1, 2, 0, 4><<<gblocks(DINNER, NTOK, 4), 256, TDM_LDS, stream>>>(
            w3bf, dtbf, delta, db, nullptr, DINNER, DTRANK, NTOK);

        scan_kernel<<<scanBlocks, 256, 0, stream>>>(
            delta, xc, xdbl, al, dpp, xz, ysum, ybf, dir, dir);
    }

    // out = out_proj_w * y + b  -> output #1; I=1024 -> MT=4
    cvt(out_proj_w, w4bf, DMODEL * DINNER);
    gemm_bf16_kernel<1, 1, 0, 4><<<gblocks(DMODEL, NTOK, 4), 256, TDM_LDS, stream>>>(
        w4bf, ybf, out1, out_proj_b, nullptr, DMODEL, DINNER, NTOK);
}
